// CostVolumeV2_58153857188503
// MI455X (gfx1250) — compile-verified, matrix operand order checked by reading, not run on hardware
//
#include <hip/hip_runtime.h>

typedef _Float16 v16h __attribute__((ext_vector_type(16)));
typedef _Float16 h8  __attribute__((ext_vector_type(8)));
typedef float    v8f __attribute__((ext_vector_type(8)));

#define B_    4
#define C_    128
#define H_    128
#define W_    256
#define ND    9          // 2*4+1 displacements per axis
#define KOUT  81
#define WX    64         // x columns per workgroup
#define NW    4          // waves per workgroup
#define CSTR  136        // padded K stride in f16 units (mult of 8 -> 16B aligned b128 frags)
#define LB_COLS 80       // nxt window cols incl. tail pad for wave 3's second N-tile
#define LA_COLS 64
#define SC_STR 33        // per-wave R scratch row stride (floats)

static __device__ __forceinline__ v16h frag16(const _Float16* p) {
    // two ds_load_b128 halves -> one 16 x f16 WMMA fragment
    h8 lo = *(const h8*)(p);
    h8 hi = *(const h8*)(p + 16);
    return __builtin_shufflevector(lo, hi, 0,1,2,3,4,5,6,7,8,9,10,11,12,13,14,15);
}

__global__ __launch_bounds__(128) void corr_wmma_kernel(
        const float* __restrict__ prv, const float* __restrict__ nxt,
        float* __restrict__ out) {
    __shared__ _Float16 __attribute__((aligned(16))) smF[LB_COLS * CSTR * 2];
    __shared__ float scb[NW * 16 * SC_STR];

    const int tid  = threadIdx.x;
    const int lane = tid & 31;
    const int wv   = tid >> 5;
    const int x0   = blockIdx.x * WX;
    const int y    = blockIdx.y;
    const int b    = blockIdx.z;

    // Time-multiplexed LDS views: A tile first, then reused as the B window.
    _Float16* lAhi = smF;
    _Float16* lAlo = smF + LA_COLS * CSTR;
    _Float16* lBhi = smF;
    _Float16* lBlo = smF + LB_COLS * CSTR;
    float* sc = scb + wv * 16 * SC_STR;

    // ---- stage prv tile (A): [col][k], f16 hi/lo split of fp32 ----
    for (int i = tid; i < LA_COLS * C_; i += NW * 32) {
        int j = i & (LA_COLS - 1);
        int c = i >> 6;
        float v = prv[((b * C_ + c) * H_ + y) * W_ + (x0 + j)];
        _Float16 hi = (_Float16)v;
        _Float16 lo = (_Float16)(v - (float)hi);
        lAhi[j * CSTR + c] = hi;
        lAlo[j * CSTR + c] = lo;
    }
    __syncthreads();

    // ---- preload A fragments (per wave: its 16 output columns; reused for all 9 dy) ----
    const int m16   = lane & 15;
    const int khalf = (lane < 16) ? 0 : 8;   // lanes<16: K 0..7 & 16..23 ; lanes>=16: K 8..15 & 24..31
    const int acol  = wv * 16 + m16;
    v16h Ahi[4], Alo[4];
#pragma unroll
    for (int kk = 0; kk < 4; ++kk) {
        int base = acol * CSTR + kk * 32 + khalf;
        Ahi[kk] = frag16(lAhi + base);
        Alo[kk] = frag16(lAlo + base);
    }

    const float invC = 1.0f / (float)C_;

    for (int dy = 0; dy < ND; ++dy) {
        const int yp = y + dy - 4;
        const bool valid = (yp >= 0) && (yp < H_);   // uniform across the WG

        __syncthreads();   // prior readers of lB done (also orders A-frag loads vs. lB stores)
        if (valid) {
            // stage 72-wide shifted nxt row window: [wincol][k], hi/lo split
            for (int i = tid; i < 72 * C_; i += NW * 32) {
                int j = i % 72;
                int c = i / 72;
                int x = x0 - 4 + j;
                float v = (x >= 0 && x < W_)
                        ? nxt[((b * C_ + c) * H_ + yp) * W_ + x] : 0.0f;
                _Float16 hi = (_Float16)v;
                _Float16 lo = (_Float16)(v - (float)hi);
                lBhi[j * CSTR + c] = hi;
                lBlo[j * CSTR + c] = lo;
            }
        }
        __syncthreads();

        if (valid) {
            v8f acc0 = {};   // R columns jlocal 0..15
            v8f acc1 = {};   // R columns jlocal 16..31
            const int jb0 = wv * 16;
#pragma unroll
            for (int kk = 0; kk < 4; ++kk) {
                int boff = kk * 32 + khalf;
                v16h Bhi0 = frag16(lBhi + (jb0 + m16) * CSTR + boff);
                v16h Blo0 = frag16(lBlo + (jb0 + m16) * CSTR + boff);
                v16h Bhi1 = frag16(lBhi + (jb0 + 16 + m16) * CSTR + boff);
                v16h Blo1 = frag16(lBlo + (jb0 + 16 + m16) * CSTR + boff);
                // fp32-faithful f16x3: hi*hi + hi*lo + lo*hi, fp32 accumulate
                acc0 = __builtin_amdgcn_wmma_f32_16x16x32_f16(false, Ahi[kk], false, Bhi0, (short)0, acc0, false, false);
                acc0 = __builtin_amdgcn_wmma_f32_16x16x32_f16(false, Ahi[kk], false, Blo0, (short)0, acc0, false, false);
                acc0 = __builtin_amdgcn_wmma_f32_16x16x32_f16(false, Alo[kk], false, Bhi0, (short)0, acc0, false, false);
                acc1 = __builtin_amdgcn_wmma_f32_16x16x32_f16(false, Ahi[kk], false, Bhi1, (short)0, acc1, false, false);
                acc1 = __builtin_amdgcn_wmma_f32_16x16x32_f16(false, Ahi[kk], false, Blo1, (short)0, acc1, false, false);
                acc1 = __builtin_amdgcn_wmma_f32_16x16x32_f16(false, Alo[kk], false, Bhi1, (short)0, acc1, false, false);
            }
            // spill R to per-wave LDS scratch for banded-diagonal extraction
            const int mrow = khalf;   // lanes<16 hold M 0..7, lanes>=16 hold M 8..15
#pragma unroll
            for (int i = 0; i < 8; ++i) {
                sc[(mrow + i) * SC_STR + m16]      = acc0[i];
                sc[(mrow + i) * SC_STR + 16 + m16] = acc1[i];
            }
            asm volatile("s_wait_dscnt 0x0" ::: "memory");   // wave-local LDS RAW
        }

        // extract the 9 diagonals: out[m,dx] = R[m, m+dx+4]; coalesced 16-wide stores
#pragma unroll
        for (int it = 0; it < 5; ++it) {
            int e = it * 32 + lane;
            if (e < 144) {
                int dxi = e >> 4;    // 0..8
                int m   = e & 15;
                float v = valid ? sc[m * SC_STR + (m + dxi)] * invC : 0.0f;
                int kout = dy * ND + dxi;
                out[((b * KOUT + kout) * H_ + y) * W_ + (x0 + wv * 16 + m)] = v;
            }
        }
    }
}

extern "C" void kernel_launch(void* const* d_in, const int* in_sizes, int n_in,
                              void* d_out, int out_size, void* d_ws, size_t ws_size,
                              hipStream_t stream) {
    (void)in_sizes; (void)n_in; (void)out_size; (void)d_ws; (void)ws_size;
    const float* prv = (const float*)d_in[0];
    const float* nxt = (const float*)d_in[1];
    float* out = (float*)d_out;
    dim3 grid(W_ / WX, H_, B_);   // (4, 128, 4) workgroups
    corr_wmma_kernel<<<grid, dim3(NW * 32), 0, stream>>>(prv, nxt, out);
}
